// CLIPLoss_22359599743077
// MI455X (gfx1250) — matrix operand order repaired and checked by reading, not compiled, not run on hardware
//
#include <hip/hip_runtime.h>

// CLIP loss, N=8192, D=1024, f32 inputs -> scalar f32 loss.
// cvt f32->f16 (scale folded into image copy), fused dual-direction WMMA-GEMM
// with ASYNC global->LDS staging + lane-local online logsumexp (cross-lane
// merge once at end), f32 diagonal, final merge/reduce. Workspace ~= 33.6 MB.

#define CN 8192
#define CD 1024
#define BM 256          // rows per workgroup (8 waves x 32 rows)
#define BN 64           // columns per block iteration (4 N-tiles)
#define BK 32           // k-chunk
#define LDT (BK + 8)    // f16 LDS row stride: 80 B, 16B-aligned, conflict-free
#define CSPLIT 8        // column splits (grid.y)

typedef __attribute__((ext_vector_type(16))) _Float16 v16h;
typedef __attribute__((ext_vector_type(8)))  _Float16 v8h;
typedef __attribute__((ext_vector_type(8)))  float    v8f;

// Direct global->LDS async copy (CDNA5 GLOBAL_LOAD_ASYNC_TO_LDS_B128, ASYNCcnt).
// VDST carries the per-lane LDS byte address (low 32 bits of the flat pointer,
// per the LDS aperture rule), VADDR the 64-bit global address (GV mode).
#define ASYNC_LD16(ldsPtr, gPtr)                                         \
  asm volatile("global_load_async_to_lds_b128 %0, %1, off"               \
               :: "v"((unsigned)(size_t)(ldsPtr)), "v"(gPtr) : "memory")
#define WAIT_ASYNC() asm volatile("s_wait_asynccnt 0x0" ::: "memory")

__global__ void clip_cvt_f16_22359(const float* __restrict__ x,
                                   _Float16* __restrict__ y,
                                   const float* __restrict__ scalep,
                                   int n, int doScale) {
  const float sc = doScale ? *scalep : 1.0f;
  int i = blockIdx.x * blockDim.x + threadIdx.x;
  int stride = gridDim.x * blockDim.x;
  for (; i < n; i += stride) y[i] = (_Float16)(x[i] * sc);
}

// diag[i] = scale * dot(img_i, txt_i), full f32 precision, one wave per row
__global__ void clip_diag_22359(const float* __restrict__ img,
                                const float* __restrict__ txt,
                                const float* __restrict__ scalep,
                                float* __restrict__ dg, int n, int d) {
  int w = (blockIdx.x * blockDim.x + threadIdx.x) >> 5;
  int lane = threadIdx.x & 31;
  if (w >= n) return;
  const float* a = img + (size_t)w * d;
  const float* b = txt + (size_t)w * d;
  float s = 0.f;
  for (int k = lane; k < d; k += 32) s += a[k] * b[k];
#pragma unroll
  for (int m = 16; m >= 1; m >>= 1) s += __shfl_xor(s, m, 32);
  if (lane == 0) dg[w] = (*scalep) * s;
}

// Fused GEMM(A * B^T) + online row logsumexp over a column split.
// grid = (N/BM, CSPLIT, 2); blockIdx.z selects direction (i2t vs t2i).
// 256 threads = 8 waves; wave w owns rows [blockIdx.x*256 + w*32, +32)
// (two 16-row M-tiles, each B fragment reused twice). Scale is pre-folded
// into the image operand, so logits come straight out of the accumulators.
__global__ __launch_bounds__(256)
void clip_gemm_lse_22359(const _Float16* __restrict__ imgH,
                         const _Float16* __restrict__ txtH,
                         float* __restrict__ pmax1, float* __restrict__ psum1,
                         float* __restrict__ pmax2, float* __restrict__ psum2,
                         int n, int d) {
  __shared__ alignas(16) _Float16 sA[2][BM * LDT];  // 40960 B
  __shared__ alignas(16) _Float16 sB[2][BN * LDT];  // 10240 B

  const int dir = blockIdx.z;
  const _Float16* __restrict__ A = dir ? txtH : imgH;
  const _Float16* __restrict__ B = dir ? imgH : txtH;
  float* __restrict__ pmax = dir ? pmax2 : pmax1;
  float* __restrict__ psum = dir ? psum2 : psum1;

  const int tid  = threadIdx.x;
  const int wave = tid >> 5;
  const int lane = tid & 31;
  const int hl   = lane >> 4;   // lane-half selects K sub-range of fragments
  const int l15  = lane & 15;
  const int rowBase = blockIdx.x * BM;
  const int colsPerSplit = n / CSPLIT;
  const int colLo = blockIdx.y * colsPerSplit;
  const int colHi = colLo + colsPerSplit;

  // staging map: 256 threads, 16B chunks; A covers 256x32 in 4 passes, B 64x32 in 1
  const int sr = tid >> 2;          // 0..63
  const int sc = (tid & 3) * 8;     // 0,8,16,24
  size_t aoff[4];
#pragma unroll
  for (int p = 0; p < 4; ++p) aoff[p] = (size_t)(rowBase + sr + p * 64) * d + sc;

  // lane-local online logsumexp state: 16 row-slots (2 M-tiles x 8 C-rows),
  // each covering only this lane's columns. Cross-lane merge once at end.
  float runMax[16], runSum[16];
#pragma unroll
  for (int q = 0; q < 16; ++q) { runMax[q] = -3.0e38f; runSum[q] = 0.f; }

  const int NK = d / BK;  // 32

  for (int colBase = colLo; colBase < colHi; colBase += BN) {
    const size_t boff = (size_t)(colBase + sr) * d + sc;

    // async-stage k-chunk 0 into buffer 0
#pragma unroll
    for (int p = 0; p < 4; ++p)
      ASYNC_LD16(&sA[0][(sr + p * 64) * LDT + sc], &A[aoff[p]]);
    ASYNC_LD16(&sB[0][sr * LDT + sc], &B[boff]);
    WAIT_ASYNC();
    __syncthreads();

    v8f acc[2][4] = {};

    for (int kk = 0; kk < NK; ++kk) {
      if (kk + 1 < NK) {
        const int kb = (kk + 1) * BK;
        const int nb = (kk + 1) & 1;
#pragma unroll
        for (int p = 0; p < 4; ++p)
          ASYNC_LD16(&sA[nb][(sr + p * 64) * LDT + sc], &A[aoff[p] + kb]);
        ASYNC_LD16(&sB[nb][sr * LDT + sc], &B[boff + kb]);
        if (kk + 2 < NK) {          // global_prefetch_b8 for the k+2 chunk
          __builtin_prefetch(&A[aoff[0] + kb + BK], 0, 1);
          __builtin_prefetch(&A[aoff[2] + kb + BK], 0, 1);
          __builtin_prefetch(&B[boff + kb + BK], 0, 1);
        }
      }
      const _Float16* pA = sA[kk & 1];
      const _Float16* pB = sB[kk & 1];

      // Preload ALL fragments first so DS latency overlaps the WMMA pipe.
      // A: lane = M row; half0: K 0-7 & 16-23, half1: K 8-15 & 24-31
      v16h af[2];
#pragma unroll
      for (int mt = 0; mt < 2; ++mt) {
        const int arow = wave * 32 + mt * 16 + l15;
        v8h alo = *(const v8h*)&pA[arow * LDT + hl * 8];
        v8h ahi = *(const v8h*)&pA[arow * LDT + 16 + hl * 8];
        af[mt] = __builtin_shufflevector(alo, ahi,
            0,1,2,3,4,5,6,7,8,9,10,11,12,13,14,15);
      }
      // B: lane = N col; half0: K 0-7 & 16-23, half1: K 8-15 & 24-31
      v16h bf[4];
#pragma unroll
      for (int t = 0; t < 4; ++t) {
        const int bcol = t * 16 + l15;
        v8h blo = *(const v8h*)&pB[bcol * LDT + hl * 8];
        v8h bhi = *(const v8h*)&pB[bcol * LDT + 16 + hl * 8];
        bf[t] = __builtin_shufflevector(blo, bhi,
            0,1,2,3,4,5,6,7,8,9,10,11,12,13,14,15);
      }
      // 8 back-to-back WMMAs; each B fragment reused for both M-tiles.
#pragma unroll
      for (int t = 0; t < 4; ++t) {
        acc[0][t] = __builtin_amdgcn_wmma_f32_16x16x32_f16(
            false, af[0], false, bf[t], (short)0, acc[0][t], false, false);
        acc[1][t] = __builtin_amdgcn_wmma_f32_16x16x32_f16(
            false, af[1], false, bf[t], (short)0, acc[1][t], false, false);
      }
      WAIT_ASYNC();       // staged chunk kk+1 fully landed in LDS
      __syncthreads();
    }

    // lane-local online LSE update (no cross-lane traffic in the hot loop)
#pragma unroll
    for (int mt = 0; mt < 2; ++mt) {
#pragma unroll
      for (int r = 0; r < 8; ++r) {
        const int q = mt * 8 + r;
        float v0 = acc[mt][0][r], v1 = acc[mt][1][r];
        float v2 = acc[mt][2][r], v3 = acc[mt][3][r];
        float bm = fmaxf(fmaxf(v0, v1), fmaxf(v2, v3));
        float nm = fmaxf(runMax[q], bm);
        runSum[q] = runSum[q] * __expf(runMax[q] - nm)
                  + __expf(v0 - nm) + __expf(v1 - nm)
                  + __expf(v2 - nm) + __expf(v3 - nm);
        runMax[q] = nm;
      }
    }
  }

  // single cross-lane (max,sum) merge over the 16 lanes of each half
#pragma unroll
  for (int q = 0; q < 16; ++q) {
    float m = runMax[q], s = runSum[q];
#pragma unroll
    for (int msk = 1; msk <= 8; msk <<= 1) {
      float om = __shfl_xor(m, msk, 32);
      float os = __shfl_xor(s, msk, 32);
      float nm = fmaxf(m, om);
      s = s * __expf(m - nm) + os * __expf(om - nm);
      m = nm;
    }
    runMax[q] = m; runSum[q] = s;
  }

  if (l15 == 0) {
#pragma unroll
    for (int mt = 0; mt < 2; ++mt) {
#pragma unroll
      for (int r = 0; r < 8; ++r) {
        const int gRow = rowBase + wave * 32 + mt * 16 + hl * 8 + r;
        pmax[(size_t)blockIdx.y * n + gRow] = runMax[mt * 8 + r];
        psum[(size_t)blockIdx.y * n + gRow] = runSum[mt * 8 + r];
      }
    }
  }
}

// Merge column splits, form both losses, reduce to scalar.
__global__ void clip_finish_22359(const float* __restrict__ pmax1,
                                  const float* __restrict__ psum1,
                                  const float* __restrict__ pmax2,
                                  const float* __restrict__ psum2,
                                  const float* __restrict__ dg,
                                  float* __restrict__ out, int n) {
  __shared__ float red[256];
  float acc = 0.f;
  for (int i = threadIdx.x; i < n; i += blockDim.x) {
    float m1 = -3.0e38f, m2 = -3.0e38f;
#pragma unroll
    for (int s = 0; s < CSPLIT; ++s) {
      m1 = fmaxf(m1, pmax1[s * n + i]);
      m2 = fmaxf(m2, pmax2[s * n + i]);
    }
    float s1 = 0.f, s2 = 0.f;
#pragma unroll
    for (int s = 0; s < CSPLIT; ++s) {
      s1 += psum1[s * n + i] * __expf(pmax1[s * n + i] - m1);
      s2 += psum2[s * n + i] * __expf(pmax2[s * n + i] - m2);
    }
    acc += (m1 + __logf(s1) - dg[i]) + (m2 + __logf(s2) - dg[i]);
  }
  red[threadIdx.x] = acc;
  __syncthreads();
  for (int o = 128; o > 0; o >>= 1) {
    if ((int)threadIdx.x < o) red[threadIdx.x] += red[threadIdx.x + o];
    __syncthreads();
  }
  if (threadIdx.x == 0) out[0] = 0.5f * red[0] / (float)n;
}

extern "C" void kernel_launch(void* const* d_in, const int* in_sizes, int n_in,
                              void* d_out, int out_size, void* d_ws, size_t ws_size,
                              hipStream_t stream) {
  (void)in_sizes; (void)n_in; (void)out_size; (void)ws_size;
  const float* img    = (const float*)d_in[0];
  const float* txt    = (const float*)d_in[1];
  const float* scalep = (const float*)d_in[2];
  float* out = (float*)d_out;
  const int N = CN, D = CD;

  // workspace layout (~33.6 MB)
  _Float16* imgH = (_Float16*)d_ws;       // image features * scale, f16
  _Float16* txtH = imgH + (size_t)N * D;  // text features, f16
  float* fws   = (float*)(txtH + (size_t)N * D);
  float* pmax1 = fws;
  float* psum1 = pmax1 + (size_t)CSPLIT * N;
  float* pmax2 = psum1 + (size_t)CSPLIT * N;
  float* psum2 = pmax2 + (size_t)CSPLIT * N;
  float* dg    = psum2 + (size_t)CSPLIT * N;

  const int total = N * D;
  clip_cvt_f16_22359<<<1024, 256, 0, stream>>>(img, imgH, scalep, total, 1);
  clip_cvt_f16_22359<<<1024, 256, 0, stream>>>(txt, txtH, scalep, total, 0);
  clip_diag_22359<<<N / 8, 256, 0, stream>>>(img, txt, scalep, dg, N, D);

  // both directions in one launch: blockIdx.z = 0 -> i2t, 1 -> t2i
  dim3 grid(N / BM, CSPLIT, 2);
  clip_gemm_lse_22359<<<grid, 256, 0, stream>>>(imgH, txtH, pmax1, psum1,
                                                pmax2, psum2, N, D);

  clip_finish_22359<<<1, 256, 0, stream>>>(pmax1, psum1, pmax2, psum2, dg, out, N);
}
